// QuerySpecificAttentionClusteringModel_9131100471909
// MI455X (gfx1250) — matrix operand order repaired and checked by reading, not compiled, not run on hardware
//
#include <hip/hip_runtime.h>
#include <stdint.h>

// ---------------------------------------------------------------------------
// Problem constants (match reference)
// ---------------------------------------------------------------------------
#define NTOK   2048
#define DIM    768
#define NHEAD  12
#define HDIM   64
#define KCLU   64
#define TEMP_INV   2.0f      // 1 / TEMP (TEMP = 0.5)
#define THRESHOLD  1e-4f
#define MAX_ITER   100
#define DKM_EPS    1e-6f
#define LN_EPS     1e-5f

// ---------------------------------------------------------------------------
// CDNA5 wave32 WMMA types
// ---------------------------------------------------------------------------
typedef __attribute__((ext_vector_type(16))) __bf16 v16bf;
typedef __attribute__((ext_vector_type(8)))  float  v8f;
typedef __attribute__((ext_vector_type(4)))  unsigned int u32x4;
typedef __attribute__((ext_vector_type(8)))  int  i32x8;
typedef __attribute__((ext_vector_type(4)))  int  i32x4;
typedef int v4i __attribute__((vector_size(16)));   // GCC-style, builtin pointee

union AFrag { v16bf v; u32x4 q[2]; };

// Feature probes (device pass only; host pass takes fallbacks, never executed)
#if defined(__gfx1250__) && __has_builtin(__builtin_amdgcn_global_load_async_to_lds_b128)
#define HAVE_ASYNC 1
#else
#define HAVE_ASYNC 0
#endif
#if defined(__gfx1250__) && __has_builtin(__builtin_amdgcn_tensor_load_to_lds)
#define HAVE_TDM 1
#else
#define HAVE_TDM 0
#endif

__device__ __forceinline__ __bf16 f2bf(float f) {
  // round-to-nearest-even fp32 -> bf16
  unsigned u = __float_as_uint(f);
  u += 0x7FFFu + ((u >> 16) & 1u);
  unsigned short h = (unsigned short)(u >> 16);
  __bf16 r;
  __builtin_memcpy(&r, &h, sizeof(r));
  return r;
}

// 16B global -> LDS copy. Async (ASYNCcnt-tracked, no VGPR bounce) when the
// gfx1250 builtin exists; otherwise synchronous load+store fallback.
// NOTE: flat addresses of __shared__ objects keep the LDS byte offset in
// addr[31:0] (ISA 10.2 aperture layout), so inttoptr into AS(3)/AS(1) is safe.
__device__ __forceinline__ void copy16(__bf16* lds, const __bf16* g) {
#if HAVE_ASYNC
  __builtin_amdgcn_global_load_async_to_lds_b128(
      (__attribute__((address_space(1))) v4i*)(uintptr_t)g,
      (__attribute__((address_space(3))) v4i*)(uintptr_t)lds, 0, 0);
#else
  *(u32x4*)lds = *(const u32x4*)g;
#endif
}

__device__ __forceinline__ void wait_async_lds() {
#if HAVE_ASYNC
#if __has_builtin(__builtin_amdgcn_s_wait_asynccnt)
  __builtin_amdgcn_s_wait_asynccnt(0);
#else
  asm volatile("s_wait_asynccnt 0x0" ::: "memory");
#endif
#endif
}

#if HAVE_TDM
// Tensor Data Mover: one DMA of a [rows x colsElems] bf16 tile (row stride
// strideElems) from global into LDS. D# per ISA 8.3/8.4: group0 = count=1,
// lds_addr, global_addr, type=2; group1 = data_size=2B, dims/strides.
__device__ __forceinline__ void tdm_load_tile_bf16(
    const __bf16* gsrc, const __bf16* ldst, int rows, int colsElems,
    int strideElems) {
  unsigned long long ga = (unsigned long long)(uintptr_t)gsrc;
  unsigned ldsOff = (unsigned)(uintptr_t)ldst;   // addr[31:0] = LDS byte offset
  unsigned td0 = 1u << 30;                       // huge tensor dims: no OOB clip
  unsigned td1 = 1u << 20;
  unsigned long long st0 = (unsigned long long)strideElems;
  u32x4 g0;
  g0[0] = 1u;                                    // count=1, user mode
  g0[1] = ldsOff;                                // lds_addr
  g0[2] = (unsigned)ga;                          // global_addr[31:0]
  g0[3] = (unsigned)((ga >> 32) & 0x01FFFFFFu) | (2u << 30);  // addr[56:32]|type=2
  i32x8 g1;
  g1[0] = 0x00010000;                            // data_size=1 (2 bytes)
  g1[1] = (int)(td0 << 16);                                  // tensor_dim0[15:0]
  g1[2] = (int)((td0 >> 16) | ((td1 & 0xFFFFu) << 16));      // td0 hi | td1 lo
  g1[3] = (int)(((td1 >> 16) & 0xFFFFu) | ((unsigned)colsElems << 16)); // tile_dim0
  g1[4] = (int)((unsigned)rows & 0xFFFFu);                   // tile_dim1 (tile_dim2=0)
  g1[5] = (int)(unsigned)(st0 & 0xFFFFFFFFu);                // dim0_stride lo
  g1[6] = (int)(unsigned)((st0 >> 32) & 0xFFFFu);            // dim0_stride hi
  g1[7] = 0;
  i32x4 z; z[0] = 0; z[1] = 0; z[2] = 0; z[3] = 0;
  i32x8 z8; z8[0] = 0; z8[1] = 0; z8[2] = 0; z8[3] = 0;
  z8[4] = 0; z8[5] = 0; z8[6] = 0; z8[7] = 0;
  __builtin_amdgcn_tensor_load_to_lds(g0, g1, z, z, z8, 0);
}
__device__ __forceinline__ void wait_tensor() {
#if __has_builtin(__builtin_amdgcn_s_wait_tensorcnt)
  __builtin_amdgcn_s_wait_tensorcnt(0);
#else
  asm volatile("s_wait_tensorcnt 0x0" ::: "memory");
#endif
}
#endif

// ---------------------------------------------------------------------------
// Tiled bf16 WMMA GEMM:  C[M,N] = A[M,Kd] * Bt[N,Kd]^T  (+bias, optional relu)
//   A  : bf16 row-major [M, Kd], lda
//   Bt : bf16 row-major [N, Kd] (column j of B == row j of Bt), ldb
//   C  : fp32 row-major [M, N],  ldc
// Block tile 64 x BN (BN in {64,128}), 2 x (BN/32) waves, each wave a 32x32
// output tile = 2x2 WMMA 16x16x32 tiles. K stepped by 32 with DOUBLE-BUFFERED
// LDS: A tile via TDM (wave 0, TENSORcnt), B tile via per-lane async loads
// (ASYNCcnt); tile k+1 is in flight while tile k feeds the WMMA pipe.
// ---------------------------------------------------------------------------
template<int BN>
__global__ __launch_bounds__(BN * 2) void gemm_bf16(
    const __bf16* __restrict__ A, int lda,
    const __bf16* __restrict__ Bt, int ldb,
    float* __restrict__ C, int ldc,
    const float* __restrict__ bias,
    int Kd, int relu, const int* __restrict__ skip)
{
  if (skip && *skip) return;            // uniform early-out for frozen DKM
  constexpr int BM    = 64;
  constexpr int NTHR  = BN * 2;
  constexpr int AITER = (BM * 4) / NTHR;   // 16B chunks per thread (A tile)
  constexpr int BITER = (BN * 4) / NTHR;   // 16B chunks per thread (B tile)
  __shared__ alignas(16) __bf16 smem[(2 * BM + 2 * BN) * 32];
  __bf16* AsB = smem;                      // [2][BM][32]
  __bf16* BsB = smem + 2 * BM * 32;        // [2][BN][32]

  const int tid  = threadIdx.x;
  const int lane = tid & 31;
  const int l15  = lane & 15;
  const int lhi  = lane >> 4;
  const int wave = tid >> 5;
  const int wm   = (wave & 1) * 32;
  const int wn   = (wave >> 1) * 32;
  const int m0   = blockIdx.x * BM;
  const int n0   = blockIdx.y * BN;

  // Hoisted per-thread staging assignment (loop-invariant rows/chunks)
  const __bf16* aSrc[AITER];
  int aDst[AITER];
  #pragma unroll
  for (int t = 0; t < AITER; ++t) {
    int c = tid + t * NTHR, r = c >> 2, qq = c & 3;
    aSrc[t] = A + (size_t)(m0 + r) * lda + qq * 8;
    aDst[t] = r * 32 + qq * 8;
  }
  const __bf16* bSrc[BITER];
  int bDst[BITER];
  #pragma unroll
  for (int t = 0; t < BITER; ++t) {
    int c = tid + t * NTHR, r = c >> 2, qq = c & 3;
    bSrc[t] = Bt + (size_t)(n0 + r) * ldb + qq * 8;
    bDst[t] = r * 32 + qq * 8;
  }

  auto stage = [&](int k0, int buf) {
#if HAVE_TDM
    if (wave == 0)        // one TDM descriptor moves the whole A tile
      tdm_load_tile_bf16(A + (size_t)m0 * lda + k0, AsB + buf * BM * 32,
                         BM, 32, lda);
#else
    #pragma unroll
    for (int t = 0; t < AITER; ++t)
      copy16(AsB + buf * BM * 32 + aDst[t], aSrc[t] + k0);
#endif
    #pragma unroll
    for (int t = 0; t < BITER; ++t)
      copy16(BsB + buf * BN * 32 + bDst[t], bSrc[t] + k0);
  };

  v8f acc[2][2];
  #pragma unroll
  for (int i = 0; i < 2; ++i)
    #pragma unroll
    for (int j = 0; j < 2; ++j)
      #pragma unroll
      for (int r = 0; r < 8; ++r) acc[i][j][r] = 0.0f;

  stage(0, 0);
  int buf = 0;
  for (int k0 = 0; k0 < Kd; k0 += 32) {
    wait_async_lds();
#if HAVE_TDM
    if (wave == 0) wait_tensor();
#endif
    __syncthreads();                      // tile `buf` visible to all waves
    if (k0 + 32 < Kd) stage(k0 + 32, buf ^ 1);   // prefetch next tile

    // A fragment (16x32): lanes 0-15 hold K 0..7/16..23, lanes 16-31 hold
    // K 8..15/24..31 for row M = lane&15 (ISA 7.12.2 16-bit A layout).
    AFrag af[2], bfr[2];
    #pragma unroll
    for (int i = 0; i < 2; ++i) {
      const __bf16* row = AsB + (buf * BM + wm + i * 16 + l15) * 32;
      af[i].q[0] = *(const u32x4*)(row + lhi * 8);
      af[i].q[1] = *(const u32x4*)(row + 16 + lhi * 8);
    }
    // B fragment (32x16): N = lane&15; lanes 0-15 cover K 0..7/16..23, lanes
    // 16-31 cover K 8..15/24..31 (same K order as A).
    #pragma unroll
    for (int j = 0; j < 2; ++j) {
      const __bf16* row = BsB + (buf * BN + wn + j * 16 + l15) * 32;
      bfr[j].q[0] = *(const u32x4*)(row + lhi * 8);
      bfr[j].q[1] = *(const u32x4*)(row + 16 + lhi * 8);
    }
    #pragma unroll
    for (int i = 0; i < 2; ++i)
      #pragma unroll
      for (int j = 0; j < 2; ++j)
        acc[i][j] = __builtin_amdgcn_wmma_f32_16x16x32_bf16(
            false, af[i].v, false, bfr[j].v, (short)0, acc[i][j], false, false);
    buf ^= 1;
  }

  // Epilogue: C/D layout — VGPR r: lanes 0-15 -> M=r, lanes 16-31 -> M=8+r,
  // N = lane&15 (ISA 7.12.2 32-bit C/D layout).
  #pragma unroll
  for (int i = 0; i < 2; ++i)
    #pragma unroll
    for (int j = 0; j < 2; ++j) {
      int col = n0 + wn + j * 16 + l15;
      float bv = bias ? bias[col] : 0.0f;
      #pragma unroll
      for (int r = 0; r < 8; ++r) {
        int row = m0 + wm + i * 16 + lhi * 8 + r;
        float v = acc[i][j][r] + bv;
        if (relu) v = fmaxf(v, 0.0f);
        C[(size_t)row * ldc + col] = v;
      }
    }
}

// ---------------------------------------------------------------------------
// Elementwise / row-wise helper kernels
// ---------------------------------------------------------------------------
__global__ void cvt_f32_bf16_kernel(const float* __restrict__ in,
                                    __bf16* __restrict__ out, int n) {
  int i = blockIdx.x * 256 + threadIdx.x;
  if (i < n) out[i] = f2bf(in[i]);
}

// qkv [N, 3D] fp32 -> Qh/Kh [H][N][HD] bf16 (q pre-scaled), Vt [H][HD][N] bf16
__global__ void repack_qkv(const float* __restrict__ qkv,
                           __bf16* __restrict__ Qh, __bf16* __restrict__ Kh,
                           __bf16* __restrict__ Vt) {
  int t = blockIdx.x * 256 + threadIdx.x;
  if (t >= NTOK * DIM) return;
  int n = t / DIM, c = t % DIM;
  int h = c / HDIM, d = c % HDIM;
  const float* row = qkv + (size_t)n * (3 * DIM);
  Qh[((size_t)h * NTOK + n) * HDIM + d] = f2bf(row[c] * 0.125f); // 1/sqrt(64)
  Kh[((size_t)h * NTOK + n) * HDIM + d] = f2bf(row[DIM + c]);
  Vt[((size_t)h * HDIM + d) * NTOK + n] = f2bf(row[2 * DIM + c]);
}

// row-wise softmax over C columns, fp32 in -> bf16 out
__global__ void softmax_rows_bf16(const float* __restrict__ in,
                                  __bf16* __restrict__ out, int C) {
  __shared__ float sh[256];
  int row = blockIdx.x, tid = threadIdx.x;
  const float* p = in + (size_t)row * C;
  __bf16* o = out + (size_t)row * C;
  float m = -3.4e38f;
  for (int i = tid; i < C; i += 256) m = fmaxf(m, p[i]);
  sh[tid] = m; __syncthreads();
  for (int s = 128; s > 0; s >>= 1) { if (tid < s) sh[tid] = fmaxf(sh[tid], sh[tid + s]); __syncthreads(); }
  m = sh[0]; __syncthreads();
  float sum = 0.f;
  for (int i = tid; i < C; i += 256) sum += __expf(p[i] - m);
  sh[tid] = sum; __syncthreads();
  for (int s = 128; s > 0; s >>= 1) { if (tid < s) sh[tid] += sh[tid + s]; __syncthreads(); }
  float inv = 1.0f / sh[0];
  for (int i = tid; i < C; i += 256) o[i] = f2bf(__expf(p[i] - m) * inv);
}

// res = E + y ; layernorm -> tr (fp32), trb (bf16), Xtb = tr^T (bf16), xsq[n]
__global__ void layernorm_kernel(const float* __restrict__ E, const float* __restrict__ y,
                                 const float* __restrict__ g, const float* __restrict__ b,
                                 float* __restrict__ tr, __bf16* __restrict__ trb,
                                 __bf16* __restrict__ Xtb, float* __restrict__ xsq) {
  __shared__ float sh[256];
  int n = blockIdx.x, tid = threadIdx.x;
  float r[3];
  float s = 0.f;
  #pragma unroll
  for (int j = 0; j < 3; ++j) {
    int c = tid + j * 256;
    r[j] = E[(size_t)n * DIM + c] + y[(size_t)n * DIM + c];
    s += r[j];
  }
  sh[tid] = s; __syncthreads();
  for (int st = 128; st > 0; st >>= 1) { if (tid < st) sh[tid] += sh[tid + st]; __syncthreads(); }
  float mu = sh[0] * (1.0f / DIM); __syncthreads();
  float v = 0.f;
  #pragma unroll
  for (int j = 0; j < 3; ++j) { float d = r[j] - mu; v += d * d; }
  sh[tid] = v; __syncthreads();
  for (int st = 128; st > 0; st >>= 1) { if (tid < st) sh[tid] += sh[tid + st]; __syncthreads(); }
  float rstd = rsqrtf(sh[0] * (1.0f / DIM) + LN_EPS); __syncthreads();
  float q = 0.f;
  #pragma unroll
  for (int j = 0; j < 3; ++j) {
    int c = tid + j * 256;
    float t = (r[j] - mu) * rstd * g[c] + b[c];
    tr[(size_t)n * DIM + c]  = t;
    trb[(size_t)n * DIM + c] = f2bf(t);
    Xtb[(size_t)c * NTOK + n] = f2bf(t);
    q += t * t;
  }
  sh[tid] = q; __syncthreads();
  for (int st = 128; st > 0; st >>= 1) { if (tid < st) sh[tid] += sh[tid + st]; __syncthreads(); }
  if (tid == 0) xsq[n] = sh[0];
}

__global__ void init_flags(int* done) { *done = 0; }

__global__ void init_centroids(const float* __restrict__ tr,
                               const long long* __restrict__ idx,
                               float* __restrict__ C) {
  int t = blockIdx.x * 256 + threadIdx.x;
  if (t >= KCLU * DIM) return;
  int k = t / DIM, d = t % DIM;
  C[t] = tr[(size_t)idx[k] * DIM + d];
}

// per-iteration prep: Cb = bf16(Ccur), csq[k] = ||C_k||^2
__global__ void dkm_prep(const float* __restrict__ C, __bf16* __restrict__ Cb,
                         float* __restrict__ csq, const int* __restrict__ done) {
  if (*done) return;
  __shared__ float sh[256];
  int k = blockIdx.x, tid = threadIdx.x;
  float s = 0.f;
  #pragma unroll
  for (int j = 0; j < 3; ++j) {
    int d = tid + j * 256;
    float v = C[(size_t)k * DIM + d];
    Cb[(size_t)k * DIM + d] = f2bf(v);
    s += v * v;
  }
  sh[tid] = s; __syncthreads();
  for (int st = 128; st > 0; st >>= 1) { if (tid < st) sh[tid] += sh[tid + st]; __syncthreads(); }
  if (tid == 0) csq[k] = sh[0];
}

// a[n,k] = softmax_k(-sqrt(max(|x|^2+|c|^2-2 x.c, 1e-12)) / TEMP); atb = a^T bf16
__global__ void dkm_assign(const float* __restrict__ dot, const float* __restrict__ xsq,
                           const float* __restrict__ csq, float* __restrict__ a,
                           __bf16* __restrict__ atb, const int* __restrict__ done) {
  if (*done) return;
  __shared__ float sh[KCLU];
  int n = blockIdx.x, k = threadIdx.x;
  float sq = xsq[n] + csq[k] - 2.0f * dot[(size_t)n * KCLU + k];
  float dist = sqrtf(fmaxf(sq, 1e-12f));
  float logit = -dist * TEMP_INV;
  sh[k] = logit; __syncthreads();
  for (int s = 32; s > 0; s >>= 1) { if (k < s) sh[k] = fmaxf(sh[k], sh[k + s]); __syncthreads(); }
  float m = sh[0]; __syncthreads();
  float e = __expf(logit - m);
  sh[k] = e; __syncthreads();
  for (int s = 32; s > 0; s >>= 1) { if (k < s) sh[k] += sh[k + s]; __syncthreads(); }
  float av = e / sh[0];
  a[(size_t)n * KCLU + k] = av;
  atb[(size_t)k * NTOK + n] = f2bf(av);
}

__global__ void dkm_asum(const float* __restrict__ a, float* __restrict__ asum,
                         const int* __restrict__ done) {
  if (*done) return;
  __shared__ float sh[256];
  int k = blockIdx.x, tid = threadIdx.x;
  float s = 0.f;
  for (int n = tid; n < NTOK; n += 256) s += a[(size_t)n * KCLU + k];
  sh[tid] = s; __syncthreads();
  for (int st = 128; st > 0; st >>= 1) { if (tid < st) sh[tid] += sh[tid + st]; __syncthreads(); }
  if (tid == 0) asum[k] = sh[0] + DKM_EPS;
}

// Cnext = Craw / asum ; pdiff[k] = sum_d |Cnext - Ccur|  (deterministic)
__global__ void dkm_update(const float* __restrict__ Craw, const float* __restrict__ asum,
                           const float* __restrict__ Ccur, float* __restrict__ Cnext,
                           float* __restrict__ pdiff, const int* __restrict__ done) {
  if (*done) return;
  __shared__ float sh[256];
  int k = blockIdx.x, tid = threadIdx.x;
  float inv = 1.0f / asum[k];
  float s = 0.f;
  #pragma unroll
  for (int j = 0; j < 3; ++j) {
    int d = tid + j * 256;
    float cn = Craw[(size_t)k * DIM + d] * inv;
    Cnext[(size_t)k * DIM + d] = cn;
    s += fabsf(cn - Ccur[(size_t)k * DIM + d]);
  }
  sh[tid] = s; __syncthreads();
  for (int st = 128; st > 0; st >>= 1) { if (tid < st) sh[tid] += sh[tid + st]; __syncthreads(); }
  if (tid == 0) pdiff[k] = sh[0];
}

__global__ void dkm_finalize(const float* __restrict__ pdiff, const float* __restrict__ Cnext,
                             float* __restrict__ Ccur, int* __restrict__ done) {
  if (*done) return;
  __shared__ float sh[64];
  int tid = threadIdx.x;
  if (tid < 64) sh[tid] = pdiff[tid];
  __syncthreads();
  for (int s = 32; s > 0; s >>= 1) { if (tid < s) sh[tid] += sh[tid + s]; __syncthreads(); }
  float total = sh[0];
  __syncthreads();
  if (total <= THRESHOLD) {
    if (tid == 0) *done = 1;             // freeze: Ccur/a are the final outputs
  } else {
    for (int i = tid; i < KCLU * DIM; i += 256) Ccur[i] = Cnext[i];
  }
}

__global__ void copy_out(const float* __restrict__ C, const float* __restrict__ a,
                         float* __restrict__ out) {
  int t = blockIdx.x * 256 + threadIdx.x;
  if (t < KCLU * DIM) out[t] = C[t];
  int t2 = t - KCLU * DIM;
  if (t2 >= 0 && t2 < NTOK * KCLU) out[KCLU * DIM + t2] = a[t2];
}

// ---------------------------------------------------------------------------
// Orchestration (all launches on `stream`; workspace arena, ~106 MB)
// ---------------------------------------------------------------------------
extern "C" void kernel_launch(void* const* d_in, const int* in_sizes, int n_in,
                              void* d_out, int out_size, void* d_ws, size_t ws_size,
                              hipStream_t stream) {
  const float* E    = (const float*)d_in[0];
  const float* fc1w = (const float*)d_in[1];
  const float* fc1b = (const float*)d_in[2];
  const float* inw  = (const float*)d_in[3];
  const float* inb  = (const float*)d_in[4];
  const float* outw = (const float*)d_in[5];
  const float* outb = (const float*)d_in[6];
  const float* fc2w = (const float*)d_in[7];
  const float* fc2b = (const float*)d_in[8];
  const float* lng  = (const float*)d_in[9];
  const float* lnb  = (const float*)d_in[10];
  const long long* idx = (const long long*)d_in[11];
  float* out = (float*)d_out;

  char* base = (char*)d_ws;
  size_t off = 0;
  auto alloc = [&](size_t bytes) -> void* {
    void* p = base + off;
    off += (bytes + 255) & ~(size_t)255;
    return p;
  };
  const size_t ND = (size_t)NTOK * DIM;

  __bf16* Eb    = (__bf16*)alloc(ND * 2);
  __bf16* W1b   = (__bf16*)alloc((size_t)DIM * DIM * 2);
  __bf16* Winb  = (__bf16*)alloc((size_t)3 * DIM * DIM * 2);
  __bf16* Woutb = (__bf16*)alloc((size_t)DIM * DIM * 2);
  __bf16* W2b   = (__bf16*)alloc((size_t)DIM * DIM * 2);
  float*  xf    = (float*)alloc(ND * 4);
  __bf16* Xb    = (__bf16*)alloc(ND * 2);
  float*  qkv   = (float*)alloc((size_t)NTOK * 3 * DIM * 4);
  __bf16* Qh    = (__bf16*)alloc((size_t)NHEAD * NTOK * HDIM * 2);
  __bf16* Kh    = (__bf16*)alloc((size_t)NHEAD * NTOK * HDIM * 2);
  __bf16* Vt    = (__bf16*)alloc((size_t)NHEAD * HDIM * NTOK * 2);
  float*  scores= (float*)alloc((size_t)NTOK * NTOK * 4);
  __bf16* probsb= (__bf16*)alloc((size_t)NTOK * NTOK * 2);
  float*  ctx   = (float*)alloc(ND * 4);
  __bf16* ctxb  = (__bf16*)alloc(ND * 2);
  float*  attn  = (float*)alloc(ND * 4);
  __bf16* attnb = (__bf16*)alloc(ND * 2);
  float*  yf    = (float*)alloc(ND * 4);
  float*  tr    = (float*)alloc(ND * 4);
  __bf16* trb   = (__bf16*)alloc(ND * 2);
  __bf16* Xtb   = (__bf16*)alloc(ND * 2);
  float*  xsq   = (float*)alloc((size_t)NTOK * 4);
  float*  Ccur  = (float*)alloc((size_t)KCLU * DIM * 4);
  float*  Cnext = (float*)alloc((size_t)KCLU * DIM * 4);
  float*  Craw  = (float*)alloc((size_t)KCLU * DIM * 4);
  __bf16* Cb    = (__bf16*)alloc((size_t)KCLU * DIM * 2);
  float*  csq   = (float*)alloc(KCLU * 4);
  float*  abuf  = (float*)alloc((size_t)NTOK * KCLU * 4);
  __bf16* atb   = (__bf16*)alloc((size_t)KCLU * NTOK * 2);
  float*  dotXC = (float*)alloc((size_t)NTOK * KCLU * 4);
  float*  asum  = (float*)alloc(KCLU * 4);
  float*  pdiff = (float*)alloc(KCLU * 4);
  int*    done  = (int*)alloc(256);

  auto cvt = [&](const float* s, __bf16* d, int n) {
    cvt_f32_bf16_kernel<<<(n + 255) / 256, 256, 0, stream>>>(s, d, n);
  };

  init_flags<<<1, 1, 0, stream>>>(done);

  // fp32 -> bf16 operands
  cvt(E, Eb, (int)ND);
  cvt(fc1w, W1b, DIM * DIM);
  cvt(inw, Winb, 3 * DIM * DIM);
  cvt(outw, Woutb, DIM * DIM);
  cvt(fc2w, W2b, DIM * DIM);

  // x = relu(E @ fc1_w^T + b)
  gemm_bf16<128><<<dim3(NTOK / 64, DIM / 128), 256, 0, stream>>>(
      Eb, DIM, W1b, DIM, xf, DIM, fc1b, DIM, 1, nullptr);
  cvt(xf, Xb, (int)ND);

  // qkv = x @ in_w^T + b
  gemm_bf16<128><<<dim3(NTOK / 64, 3 * DIM / 128), 256, 0, stream>>>(
      Xb, DIM, Winb, DIM, qkv, 3 * DIM, inb, DIM, 0, nullptr);
  repack_qkv<<<(int)((ND + 255) / 256), 256, 0, stream>>>(qkv, Qh, Kh, Vt);

  // attention, one head at a time (scores buffer reuse)
  for (int h = 0; h < NHEAD; ++h) {
    const __bf16* qh = Qh + (size_t)h * NTOK * HDIM;
    const __bf16* kh = Kh + (size_t)h * NTOK * HDIM;
    const __bf16* vt = Vt + (size_t)h * HDIM * NTOK;
    gemm_bf16<128><<<dim3(NTOK / 64, NTOK / 128), 256, 0, stream>>>(
        qh, HDIM, kh, HDIM, scores, NTOK, nullptr, HDIM, 0, nullptr);
    softmax_rows_bf16<<<NTOK, 256, 0, stream>>>(scores, probsb, NTOK);
    gemm_bf16<64><<<dim3(NTOK / 64, 1), 128, 0, stream>>>(
        probsb, NTOK, vt, NTOK, ctx + h * HDIM, DIM, nullptr, NTOK, 0, nullptr);
  }
  cvt(ctx, ctxb, (int)ND);

  // attn = ctx @ out_w^T + b ; y = relu(attn @ fc2_w^T + b)
  gemm_bf16<128><<<dim3(NTOK / 64, DIM / 128), 256, 0, stream>>>(
      ctxb, DIM, Woutb, DIM, attn, DIM, outb, DIM, 0, nullptr);
  cvt(attn, attnb, (int)ND);
  gemm_bf16<128><<<dim3(NTOK / 64, DIM / 128), 256, 0, stream>>>(
      attnb, DIM, W2b, DIM, yf, DIM, fc2b, DIM, 1, nullptr);

  // layernorm(E + y) -> tr / trb / Xtb / xsq
  layernorm_kernel<<<NTOK, 256, 0, stream>>>(E, yf, lng, lnb, tr, trb, Xtb, xsq);

  // DKM init + fixed 1 + MAX_ITER assignment rounds with done-freeze
  init_centroids<<<(KCLU * DIM + 255) / 256, 256, 0, stream>>>(tr, idx, Ccur);
  for (int it = 0; it <= MAX_ITER; ++it) {
    dkm_prep<<<KCLU, 256, 0, stream>>>(Ccur, Cb, csq, done);
    gemm_bf16<64><<<dim3(NTOK / 64, 1), 128, 0, stream>>>(
        trb, DIM, Cb, DIM, dotXC, KCLU, nullptr, DIM, 0, done);
    dkm_assign<<<NTOK, KCLU, 0, stream>>>(dotXC, xsq, csq, abuf, atb, done);
    dkm_asum<<<KCLU, 256, 0, stream>>>(abuf, asum, done);
    gemm_bf16<128><<<dim3(1, DIM / 128), 256, 0, stream>>>(
        atb, NTOK, Xtb, NTOK, Craw, DIM, nullptr, NTOK, 0, done);
    dkm_update<<<KCLU, 256, 0, stream>>>(Craw, asum, Ccur, Cnext, pdiff, done);
    dkm_finalize<<<1, 256, 0, stream>>>(pdiff, Cnext, Ccur, done);
  }

  // outputs: C [64,768] then a [2048,64]
  copy_out<<<(KCLU * DIM + NTOK * KCLU + 255) / 256, 256, 0, stream>>>(Ccur, abuf, out);
}
